// GNNModel_57578331570510
// MI455X (gfx1250) — compile-verified, matrix operand order checked by reading, not run on hardware
//
#include <hip/hip_runtime.h>
#include <hip/hip_bf16.h>
#include <cstdint>

#define B_    4
#define N_    4096
#define DIN_  128
#define DOUT_ 64
#define E_    131072
#define ED_   8
#define EF_   (E_ + N_)          // edges + self loops = 135168
#define KD_   (N_ * DOUT_)       // 262144 (dense K)
#define NEG_SLOPE 0.2f
#define NSLAB 64                 // split-K slabs for the dense GEMM

typedef __attribute__((ext_vector_type(16))) __bf16 v16bf;
typedef __attribute__((ext_vector_type(8)))  float  v8f;

static __device__ __forceinline__ __bf16 f2bf(float f) {
  unsigned u = __float_as_uint(f);
  u += 0x7fffu + ((u >> 16) & 1u);           // round-to-nearest-even
  unsigned short h = (unsigned short)(u >> 16);
  return __builtin_bit_cast(__bf16, h);
}

static __device__ __forceinline__ void atomicMaxF(float* addr, float v) {
  if (v >= 0.f) atomicMax((int*)addr, __float_as_int(v));
  else          atomicMin((unsigned int*)addr, __float_as_uint(v));
}

// ---------------------------------------------------------------- utilities
__global__ void k_fill(float* p, float v, int n) {
  int i = blockIdx.x * blockDim.x + threadIdx.x;
  if (i < n) p[i] = v;
}

// deg + per-target sum of edge_attr (for self-loop fill_value='mean')
__global__ void k_scatter_edges(const int* ei, const float* ef,
                                float* deg, float* lsum) {
  int i = blockIdx.x * blockDim.x + threadIdx.x;   // over E*8
  if (i >= E_ * ED_) return;
  int e = i >> 3, k = i & 7;
  int dst = ei[E_ + e];
  atomicAdd(&lsum[dst * ED_ + k], ef[i]);
  if (k == 0) atomicAdd(&deg[dst], 1.0f);
}

__global__ void k_loop_div(float* lsum, const float* deg) {
  int i = blockIdx.x * blockDim.x + threadIdx.x;   // over N*8
  if (i >= N_ * ED_) return;
  lsum[i] = lsum[i] / fmaxf(deg[i >> 3], 1.0f);
}

// ------------------------------------------------ WMMA 1: xl/xr projections
// grid: (N/16, B), block: 128 (4 waves); wave w -> n-tile [16w, 16w+16)
__global__ void __launch_bounds__(128)
k_xlxr_wmma(const float* __restrict__ x,
            const float* __restrict__ Wl, const float* __restrict__ bl,
            const float* __restrict__ Wr, const float* __restrict__ br,
            float* __restrict__ xl, float* __restrict__ xr) {
  const int b     = blockIdx.y;
  const int m0    = blockIdx.x * 16;
  const int wave  = threadIdx.x >> 5;
  const int l     = threadIdx.x & 31;
  const int n0    = wave * 16;
  const int mrow  = l & 15;
  const int khalf = (l >> 4) * 8;          // upper half-wave handles K+8 block
  const float* xrow = x + ((size_t)b * N_ + (m0 + mrow)) * DIN_;
  v8f cl = {}; v8f cr = {};
  for (int k0 = 0; k0 < DIN_; k0 += 32) {
    v16bf a, bL, bR;
#pragma unroll
    for (int i = 0; i < 8; ++i) {
      int kr0 = k0 + khalf + i;
      int kr1 = k0 + 16 + khalf + i;
      a[i]     = f2bf(xrow[kr0]);
      a[8 + i] = f2bf(xrow[kr1]);
      bL[i]     = f2bf(Wl[kr0 * DOUT_ + n0 + mrow]);
      bL[8 + i] = f2bf(Wl[kr1 * DOUT_ + n0 + mrow]);
      bR[i]     = f2bf(Wr[kr0 * DOUT_ + n0 + mrow]);
      bR[8 + i] = f2bf(Wr[kr1 * DOUT_ + n0 + mrow]);
    }
    cl = __builtin_amdgcn_wmma_f32_16x16x32_bf16(false, a, false, bL, (short)0, cl, false, false);
    cr = __builtin_amdgcn_wmma_f32_16x16x32_bf16(false, a, false, bR, (short)0, cr, false, false);
  }
  const int col = n0 + mrow;
#pragma unroll
  for (int r = 0; r < 8; ++r) {
    int row = m0 + r + ((l >> 4) << 3);
    size_t idx = ((size_t)b * N_ + row) * DOUT_ + col;
    xl[idx] = cl[r] + bl[col];
    xr[idx] = cr[r] + br[col];
  }
}

// ------------------------------------------------ attention, warp-per-edge
__global__ void __launch_bounds__(256)
k_edge_logits(const int* __restrict__ ei, const float* __restrict__ ef,
              const float* __restrict__ loop_attr,
              const float* __restrict__ We, const float* __restrict__ att,
              const float* __restrict__ xl, const float* __restrict__ xr,
              float* __restrict__ aout, float* __restrict__ amax) {
  __shared__ float sWe[ED_ * DOUT_];
  for (int i = threadIdx.x; i < ED_ * DOUT_; i += blockDim.x) sWe[i] = We[i];
  __syncthreads();
  const int b = blockIdx.y;
  const int warp = threadIdx.x >> 5, lane = threadIdx.x & 31;
  const int e = blockIdx.x * 8 + warp;
  if (e >= EF_) return;
  int src, dst; const float* ea;
  if (e < E_) { src = ei[e]; dst = ei[E_ + e]; ea = ef + (size_t)e * ED_; }
  else        { src = e - E_; dst = src;       ea = loop_attr + (size_t)(e - E_) * ED_; }
  float eav = (lane < ED_) ? ea[lane] : 0.f;
  float ee0 = 0.f, ee1 = 0.f;
#pragma unroll
  for (int k = 0; k < ED_; ++k) {
    float ek = __shfl(eav, k, 32);
    ee0 += ek * sWe[k * DOUT_ + lane];
    ee1 += ek * sWe[k * DOUT_ + lane + 32];
  }
  const float* xls = xl + ((size_t)b * N_ + src) * DOUT_;
  const float* xrt = xr + ((size_t)b * N_ + dst) * DOUT_;
  float m0 = xls[lane]      + xrt[lane]      + ee0;
  float m1 = xls[lane + 32] + xrt[lane + 32] + ee1;
  m0 = m0 > 0.f ? m0 : NEG_SLOPE * m0;
  m1 = m1 > 0.f ? m1 : NEG_SLOPE * m1;
  float v = m0 * att[lane] + m1 * att[lane + 32];
#pragma unroll
  for (int o = 16; o > 0; o >>= 1) v += __shfl_xor(v, o, 32);
  if (lane == 0) {
    aout[(size_t)b * EF_ + e] = v;
    atomicMaxF(&amax[b * N_ + dst], v);
  }
}

__global__ void k_edge_exp(const int* __restrict__ ei, const float* __restrict__ amax,
                           float* __restrict__ aout, float* __restrict__ denom) {
  int e = blockIdx.x * blockDim.x + threadIdx.x;
  int b = blockIdx.y;
  if (e >= EF_) return;
  int dst = (e < E_) ? ei[E_ + e] : (e - E_);
  size_t idx = (size_t)b * EF_ + e;
  float ex = expf(aout[idx] - amax[b * N_ + dst]);
  aout[idx] = ex;
  atomicAdd(&denom[b * N_ + dst], ex);
}

__global__ void __launch_bounds__(256)
k_edge_scatter(const int* __restrict__ ei, const float* __restrict__ aout,
               const float* __restrict__ denom, const float* __restrict__ xl,
               float* __restrict__ accum) {
  const int b = blockIdx.y;
  const int warp = threadIdx.x >> 5, lane = threadIdx.x & 31;
  const int e = blockIdx.x * 8 + warp;
  if (e >= EF_) return;
  int src, dst;
  if (e < E_) { src = ei[e]; dst = ei[E_ + e]; } else { src = e - E_; dst = src; }
  float w = aout[(size_t)b * EF_ + e] / (denom[b * N_ + dst] + 1e-16f);
  const float* xls = xl + ((size_t)b * N_ + src) * DOUT_;
  float* acc = accum + ((size_t)b * N_ + dst) * DOUT_;
  atomicAdd(&acc[lane],      xls[lane]      * w);
  atomicAdd(&acc[lane + 32], xls[lane + 32] * w);
}

// relu(accum + bias) -> bf16 activations for the dense layer
__global__ void k_finalize(const float* __restrict__ accum,
                           const float* __restrict__ bias, __bf16* __restrict__ Xbf) {
  size_t i = (size_t)blockIdx.x * blockDim.x + threadIdx.x;
  if (i >= (size_t)B_ * N_ * DOUT_) return;
  float y = accum[i] + bias[i % DOUT_];
  Xbf[i] = f2bf(y > 0.f ? y : 0.f);
}

// ------------------------------------------------ WMMA 2: [16,262144]x[262144,64]
// grid: NSLAB blocks of 128 (4 waves); wave w -> n-tile w; split-K + atomics
__global__ void __launch_bounds__(128)
k_dense_wmma(const __bf16* __restrict__ Xbf, const float* __restrict__ Wd,
             float* __restrict__ partial) {
  const int wave  = threadIdx.x >> 5;
  const int l     = threadIdx.x & 31;
  const int n0    = wave * 16;
  const int mrow  = l & 15;
  const int khalf = (l >> 4) * 8;
  const int col   = n0 + mrow;
  const int kbeg  = blockIdx.x * (KD_ / NSLAB);
  const int kend  = kbeg + (KD_ / NSLAB);
  const bool valid = (mrow < B_);
  const __bf16* arow = Xbf + (size_t)(mrow & 3) * KD_;  // alias rows 4..15 -> 0..3
  const __bf16 zbf = f2bf(0.f);
  v8f c = {};
  for (int k0 = kbeg; k0 < kend; k0 += 32) {
    v16bf a, bb;
#pragma unroll
    for (int i = 0; i < 8; ++i) {
      int kr0 = k0 + khalf + i;
      int kr1 = k0 + 16 + khalf + i;
      a[i]     = valid ? arow[kr0] : zbf;
      a[8 + i] = valid ? arow[kr1] : zbf;
      bb[i]     = f2bf(Wd[(size_t)kr0 * DOUT_ + col]);
      bb[8 + i] = f2bf(Wd[(size_t)kr1 * DOUT_ + col]);
    }
    c = __builtin_amdgcn_wmma_f32_16x16x32_bf16(false, a, false, bb, (short)0, c, false, false);
  }
#pragma unroll
  for (int r = 0; r < 8; ++r) {
    int row = r + ((l >> 4) << 3);
    atomicAdd(&partial[row * DOUT_ + col], c[r]);   // rows >= 4 accumulate zeros
  }
}

__global__ void k_out(const float* __restrict__ partial, const float* __restrict__ bd,
                      float* __restrict__ out) {
  int i = threadIdx.x;                       // 256 = B*DOUT
  float y = partial[i] + bd[i & 63];
  out[i] = fmaxf(y, 0.f);
}

// ---------------------------------------------------------------- launcher
extern "C" void kernel_launch(void* const* d_in, const int* in_sizes, int n_in,
                              void* d_out, int out_size, void* d_ws, size_t ws_size,
                              hipStream_t stream) {
  const float* x    = (const float*)d_in[0];
  const int*   ei   = (const int*)  d_in[1];
  const float* ef   = (const float*)d_in[2];
  const float* Wl   = (const float*)d_in[3];
  const float* bl   = (const float*)d_in[4];
  const float* Wr   = (const float*)d_in[5];
  const float* br   = (const float*)d_in[6];
  const float* We   = (const float*)d_in[7];
  const float* att  = (const float*)d_in[8];
  const float* bias = (const float*)d_in[9];
  const float* Wd   = (const float*)d_in[10];
  const float* bd   = (const float*)d_in[11];
  float* out = (float*)d_out;

  // workspace layout (float offsets)
  float* ws = (float*)d_ws;
  float* deg    = ws;                          // 4096
  float* lattr  = deg   + N_;                  // 32768 (sums, then means in place)
  float* xl     = lattr + N_ * ED_;            // B*N*64
  float* xr     = xl    + (size_t)B_ * N_ * DOUT_;
  float* aout   = xr    + (size_t)B_ * N_ * DOUT_;   // B*EF
  float* amax   = aout  + (size_t)B_ * EF_;          // B*N
  float* denom  = amax  + (size_t)B_ * N_;           // B*N
  float* accum  = denom + (size_t)B_ * N_;           // B*N*64
  float* part   = accum + (size_t)B_ * N_ * DOUT_;   // 16*64
  __bf16* Xbf   = (__bf16*)(part + 16 * DOUT_);      // B*KD bf16

  // init
  k_fill<<<(N_ + N_ * ED_ + 255) / 256, 256, 0, stream>>>(deg, 0.f, N_ + N_ * ED_);
  k_fill<<<(B_ * N_ + 255) / 256, 256, 0, stream>>>(amax, -3.402823466e38f, B_ * N_);
  {
    int ztot = B_ * N_ + B_ * N_ * DOUT_ + 16 * DOUT_;    // denom+accum+part (contiguous)
    k_fill<<<(ztot + 255) / 256, 256, 0, stream>>>(denom, 0.f, ztot);
  }

  // self-loop attr (mean of incoming edge features)
  k_scatter_edges<<<(E_ * ED_) / 256, 256, 0, stream>>>(ei, ef, deg, lattr);
  k_loop_div<<<(N_ * ED_) / 256, 256, 0, stream>>>(lattr, deg);

  // xl / xr projections (WMMA bf16)
  k_xlxr_wmma<<<dim3(N_ / 16, B_), 128, 0, stream>>>(x, Wl, bl, Wr, br, xl, xr);

  // GATv2 attention + softmax + aggregate
  dim3 eg(EF_ / 8, B_);
  k_edge_logits <<<eg, 256, 0, stream>>>(ei, ef, lattr, We, att, xl, xr, aout, amax);
  k_edge_exp    <<<dim3((EF_ + 255) / 256, B_), 256, 0, stream>>>(ei, amax, aout, denom);
  k_edge_scatter<<<eg, 256, 0, stream>>>(ei, aout, denom, xl, accum);
  k_finalize    <<<(B_ * N_ * DOUT_) / 256, 256, 0, stream>>>(accum, bias, Xbf);

  // dense head (split-K WMMA bf16) + epilogue
  k_dense_wmma<<<NSLAB, 128, 0, stream>>>(Xbf, Wd, part);
  k_out<<<1, 256, 0, stream>>>(part, bd, out);
}
